// KPAE_41549513622098
// MI455X (gfx1250) — matrix operand order repaired and checked by reading, not compiled, not run on hardware
//
#include <hip/hip_runtime.h>
#include <hip/hip_bf16.h>

typedef __attribute__((ext_vector_type(16))) __bf16 v16bf;
typedef __attribute__((ext_vector_type(8)))  float  v8f;

#define LREL 0.01f

__device__ __forceinline__ v16bf ld_v16bf(const __bf16* p) {
  union { v16bf v; uint4 q[2]; } u;
  u.q[0] = *(const uint4*)(p);
  u.q[1] = *(const uint4*)(p + 8);
  return u.v;
}

// ---------------------------------------------------------------------------
// Prep: sample means, build input_map (bf16, 32-ch padded) + copy into the
// 160-ch concat buffer at channel offset 128, and fp32 radiance mean.
// ---------------------------------------------------------------------------
__global__ void prep_kernel(const float* __restrict__ radiance,
                            const float* __restrict__ features,
                            const float* __restrict__ gfeat,
                            __bf16* __restrict__ imap,   // [B][16384][32]
                            __bf16* __restrict__ kbuf,   // [B][16384][160] (ch 128..159)
                            float*  __restrict__ rad)    // [B][3][16384]
{
  int p = blockIdx.x * blockDim.x + threadIdx.x;   // pixel 0..16383
  int b = blockIdx.y;
  if (p >= 128 * 128) return;
  size_t pix = (size_t)b * 16384 + p;
  __bf16* im = imap + pix * 32;
  __bf16* kb = kbuf + pix * 160 + 128;
  for (int c = 0; c < 20; ++c) {
    float s = 0.f;
    for (int sm = 0; sm < 8; ++sm)
      s += features[(((size_t)b * 8 + sm) * 20 + c) * 16384 + p];
    __bf16 h = (__bf16)(s * 0.125f);
    im[c] = h; kb[c] = h;
  }
  for (int c = 0; c < 4; ++c) {
    __bf16 h = (__bf16)gfeat[b * 4 + c];
    im[20 + c] = h; kb[20 + c] = h;
  }
  for (int c = 24; c < 32; ++c) { im[c] = (__bf16)0.f; kb[c] = (__bf16)0.f; }
  for (int c = 0; c < 3; ++c) {
    float s = 0.f;
    for (int sm = 0; sm < 8; ++sm)
      s += radiance[(((size_t)b * 8 + sm) * 3 + c) * 16384 + p];
    rad[((size_t)b * 3 + c) * 16384 + p] = s * 0.125f;
  }
}

// ---------------------------------------------------------------------------
// Pack OIHW fp32 weights into a WMMA bf16 A-fragment layout (contiguous k, same order as B):
// block (ct, tap, kc) = 16x32 tile; lane l holds M=l%16,
// K = j + (l>=16 ? 16 : 0) for j=0..15.
// ---------------------------------------------------------------------------
__global__ void pack_weights(const float* __restrict__ w, __bf16* __restrict__ dst,
                             int Cout, int Cin, int KC, int CT)
{
  int idx = blockIdx.x * blockDim.x + threadIdx.x;
  int total = CT * 9 * KC * 32;
  if (idx >= total) return;
  int lane = idx & 31;
  int rest = idx >> 5;
  int kc = rest % KC; rest /= KC;
  int tap = rest % 9;
  int ct  = rest / 9;
  int cout = ct * 16 + (lane & 15);
  int koff = (lane >= 16) ? 16 : 0;
  __bf16* d = dst + (size_t)idx * 16;
  for (int j = 0; j < 16; ++j) {
    int K = j + koff;
    int cin = kc * 32 + K;
    float v = 0.f;
    if (cout < Cout && cin < Cin)
      v = w[((size_t)cout * Cin + cin) * 9 + tap];
    d[j] = (__bf16)v;
  }
}

// ---------------------------------------------------------------------------
// Implicit-GEMM 3x3 conv, NHWC bf16 in/out (SIN == KC*32), f32 accumulate.
// 256 threads = 8 wave32; each wave: 2 cout tiles x 8 pixel rows so one LDS
// B-fragment feeds two WMMAs (halves LDS traffic vs 1x16 mapping).
// LDS: 18x18 halo tile, per-pixel stride KC*32+8 halves (bank-conflict-free),
// filled with async global->LDS copies (ASYNCcnt) + ds zero-fill for OOB.
// ---------------------------------------------------------------------------
template <int KC>
__global__ void __launch_bounds__(256) conv3x3_wmma(
    const __bf16* __restrict__ in, __bf16* __restrict__ out,
    const __bf16* __restrict__ wp, const float* __restrict__ bias,
    int Hin, int Win, int Hout, int Wout, int SOUT,
    int CT, int Cout, int pad, int leaky, int groups)
{
  extern __shared__ __bf16 lds[];
  constexpr int SIN  = KC * 32;
  constexpr int LDSS = KC * 32 + 8;
  const int b = blockIdx.z / groups;
  const int g = blockIdx.z % groups;
  const int oy0 = blockIdx.y * 16;
  const int ox0 = blockIdx.x * 16;
  const __bf16* inb = in + (size_t)b * Hin * Win * SIN;

  // cooperative halo-tile fill: async DMA for in-bounds, ds zero for OOB
  for (int idx = threadIdx.x; idx < 18 * 18; idx += 256) {
    int ty = idx / 18, tx = idx % 18;
    int gy = oy0 - pad + ty, gx = ox0 - pad + tx;
    __bf16* dstp = lds + (size_t)idx * LDSS;
    if (gy >= 0 && gy < Hin && gx >= 0 && gx < Win) {
      const __bf16* srcp = inb + ((size_t)gy * Win + gx) * SIN;
      unsigned ldsoff = (unsigned)(size_t)dstp;          // low 32b = LDS offset
      unsigned long long ga = (unsigned long long)(size_t)srcp;
#pragma unroll
      for (int q = 0; q < KC * 4; ++q) {
        asm volatile("global_load_async_to_lds_b128 %0, %1, off"
                     :: "v"(ldsoff + (unsigned)(q * 16)),
                        "v"(ga + (unsigned long long)(q * 16))
                     : "memory");
      }
    } else {
      uint4 z = make_uint4(0u, 0u, 0u, 0u);
      uint4* dq = (uint4*)dstp;
      for (int q = 0; q < KC * 4; ++q) dq[q] = z;
    }
  }
  asm volatile("s_wait_asynccnt 0x0" ::: "memory");
  __syncthreads();

  const int wv   = threadIdx.x >> 5;    // wave32
  const int lane = threadIdx.x & 31;
  const int ct0     = g * 8 + (wv & 3) * 2;   // this wave's two cout tiles
  const int rowbase = (wv >> 2) * 8;          // this wave's 8 output rows
  if (ct0 >= CT) return;                      // uniform per wave, after barrier

  const int mrow = lane & 15;           // B column (pixel x) / A row (cout)
  const int hi   = lane >> 4;

  v8f acc0[8], acc1[8];
#pragma unroll
  for (int n = 0; n < 8; ++n)
#pragma unroll
    for (int r = 0; r < 8; ++r) { acc0[n][r] = 0.f; acc1[n][r] = 0.f; }

  const size_t wstride = (size_t)9 * KC * 512;   // halves per cout tile
  const __bf16* wbase = wp + (size_t)ct0 * wstride + (size_t)lane * 16;

  for (int tap = 0; tap < 9; ++tap) {
    const int dy = tap / 3, dx = tap % 3;
    const __bf16* lrow = lds + (size_t)(rowbase + dy) * (18 * LDSS)
                             + (size_t)(mrow + dx) * LDSS + hi * 16;
#pragma unroll
    for (int kc = 0; kc < KC; ++kc) {
      const __bf16* ap = wbase + (size_t)(tap * KC + kc) * 512;
      v16bf a0 = ld_v16bf(ap);
      v16bf a1 = ld_v16bf(ap + wstride);
      const __bf16* lb = lrow + kc * 32;
      v16bf bcur = ld_v16bf(lb);
#pragma unroll
      for (int n = 0; n < 8; ++n) {
        v16bf bnext;
        if (n < 7) bnext = ld_v16bf(lb + (size_t)(n + 1) * (18 * LDSS));
        acc0[n] = __builtin_amdgcn_wmma_f32_16x16x32_bf16(
            false, a0, false, bcur, (short)0, acc0[n], false, false);
        acc1[n] = __builtin_amdgcn_wmma_f32_16x16x32_bf16(
            false, a1, false, bcur, (short)0, acc1[n], false, false);
        if (n < 7) bcur = bnext;
      }
    }
  }

  __bf16* outb = out + (size_t)b * Hout * Wout * SOUT;
  const int ox = ox0 + mrow;
#pragma unroll
  for (int t = 0; t < 2; ++t) {
    const int coutbase = (ct0 + t) * 16 + hi * 8;
    float bs[8];
#pragma unroll
    for (int r = 0; r < 8; ++r)
      bs[r] = (coutbase + r < Cout) ? bias[coutbase + r] : 0.f;
    if (ox < Wout) {
#pragma unroll
      for (int n = 0; n < 8; ++n) {
        int oy = oy0 + rowbase + n;
        if (oy < Hout) {
          union { uint4 q; __bf16 h[8]; } st;
#pragma unroll
          for (int r = 0; r < 8; ++r) {
            float v = (t ? acc1[n][r] : acc0[n][r]) + bs[r];
            if (leaky) v = (v > 0.f) ? v : LREL * v;
            st.h[r] = (__bf16)v;
          }
          *(uint4*)(outb + ((size_t)oy * Wout + ox) * SOUT + coutbase) = st.q;
        }
      }
    }
  }
}

// ---------------------------------------------------------------------------
// Fused softmax(441) + 21x21 kernel apply (fp32).
// ---------------------------------------------------------------------------
__global__ void apply_softmax_kernel(const float* __restrict__ rad,
                                     const __bf16* __restrict__ kcoef,
                                     float* __restrict__ out)
{
  const int HP = 108;
  int p = blockIdx.x * blockDim.x + threadIdx.x;
  int b = blockIdx.y;
  if (p >= HP * HP) return;
  int oy = p / HP, ox = p % HP;
  const __bf16* kc = kcoef + (((size_t)b * 122 + (oy + 7)) * 122 + (ox + 7)) * 448;
  float mx = -1e30f;
  for (int k = 0; k < 441; ++k) mx = fmaxf(mx, (float)kc[k]);
  const float* r0 = rad + (size_t)b * 3 * 16384;
  float s = 0.f, a0 = 0.f, a1 = 0.f, a2 = 0.f;
  int k = 0;
  for (int i = 0; i < 21; ++i) {
    int rowoff = (oy + i) * 128 + ox;
    for (int j = 0; j < 21; ++j, ++k) {
      float e = __expf((float)kc[k] - mx);
      s  += e;
      a0 += e * r0[rowoff + j];
      a1 += e * r0[16384 + rowoff + j];
      a2 += e * r0[32768 + rowoff + j];
    }
  }
  float inv = 1.f / s;
  out[((size_t)b * 3 + 0) * (HP * HP) + p] = a0 * inv;
  out[((size_t)b * 3 + 1) * (HP * HP) + p] = a1 * inv;
  out[((size_t)b * 3 + 2) * (HP * HP) + p] = a2 * inv;
}

// ---------------------------------------------------------------------------
extern "C" void kernel_launch(void* const* d_in, const int* in_sizes, int n_in,
                              void* d_out, int out_size, void* d_ws, size_t ws_size,
                              hipStream_t stream) {
  const float* radiance = (const float*)d_in[0];
  const float* features = (const float*)d_in[1];
  const float* gfeat    = (const float*)d_in[2];
  const float* mw0      = (const float*)d_in[3];
  const float* mb0      = (const float*)d_in[4];
  const float* mw_mid   = (const float*)d_in[5];
  const float* mb_mid   = (const float*)d_in[6];
  const float* kw0      = (const float*)d_in[7];
  const float* kb0      = (const float*)d_in[8];
  const float* kw1      = (const float*)d_in[9];
  const float* kb1      = (const float*)d_in[10];
  const float* kw2      = (const float*)d_in[11];
  const float* kb2      = (const float*)d_in[12];
  float* out = (float*)d_out;
  (void)in_sizes; (void)n_in; (void)out_size; (void)ws_size;

  const int B = 4;
  char* ws = (char*)d_ws;
  size_t off = 0;
  auto alloc = [&](size_t bytes) {
    char* p = ws + off;
    off = (off + bytes + 255) & ~(size_t)255;
    return p;
  };
  __bf16* imap = (__bf16*)alloc((size_t)B * 16384 * 32 * 2);
  float*  radm = (float*) alloc((size_t)B * 3 * 16384 * 4);
  __bf16* ping = (__bf16*)alloc((size_t)B * 16384 * 128 * 2);
  __bf16* pong = (__bf16*)alloc((size_t)B * 16384 * 128 * 2);
  __bf16* kbuf = (__bf16*)alloc((size_t)B * 16384 * 160 * 2);
  __bf16* k1o  = (__bf16*)alloc((size_t)B * 126 * 126 * 128 * 2);
  __bf16* k2o  = (__bf16*)alloc((size_t)B * 124 * 124 * 128 * 2);
  __bf16* kco  = (__bf16*)alloc((size_t)B * 122 * 122 * 448 * 2);
  __bf16* wp0  = (__bf16*)alloc((size_t)8  * 9 * 1 * 512 * 2);
  __bf16* wpm  = (__bf16*)alloc((size_t)11 * 8 * 9 * 4 * 512 * 2);
  __bf16* wpk0 = (__bf16*)alloc((size_t)8  * 9 * 5 * 512 * 2);
  __bf16* wpk1 = (__bf16*)alloc((size_t)8  * 9 * 4 * 512 * 2);
  __bf16* wpk2 = (__bf16*)alloc((size_t)28 * 9 * 4 * 512 * 2);

  hipFuncSetAttribute(reinterpret_cast<const void*>(conv3x3_wmma<1>),
                      hipFuncAttributeMaxDynamicSharedMemorySize, 128 * 1024);
  hipFuncSetAttribute(reinterpret_cast<const void*>(conv3x3_wmma<4>),
                      hipFuncAttributeMaxDynamicSharedMemorySize, 128 * 1024);
  hipFuncSetAttribute(reinterpret_cast<const void*>(conv3x3_wmma<5>),
                      hipFuncAttributeMaxDynamicSharedMemorySize, 128 * 1024);

  auto packs = [&](const float* w, __bf16* dst, int Cout, int Cin, int KC, int CT) {
    int total = CT * 9 * KC * 32;
    pack_weights<<<(total + 255) / 256, 256, 0, stream>>>(w, dst, Cout, Cin, KC, CT);
  };
  packs(mw0, wp0, 128, 24, 1, 8);
  for (int i = 0; i < 11; ++i)
    packs(mw_mid + (size_t)i * 128 * 128 * 9, wpm + (size_t)i * 8 * 9 * 4 * 512,
          128, 128, 4, 8);
  packs(kw0, wpk0, 128, 152, 5, 8);
  packs(kw1, wpk1, 128, 128, 4, 8);
  packs(kw2, wpk2, 441, 128, 4, 28);

  prep_kernel<<<dim3(64, B), 256, 0, stream>>>(radiance, features, gfeat,
                                               imap, kbuf, radm);

  auto conv = [&](const __bf16* ip, __bf16* op, const __bf16* wpt, const float* bias,
                  int Hin, int Win, int Hout, int Wout, int SOUT,
                  int KC, int CT, int Cout, int pad, int leaky, int groups) {
    int tX = (Wout + 15) / 16, tY = (Hout + 15) / 16;
    size_t ldsb = (size_t)18 * 18 * (KC * 32 + 8) * 2;
    dim3 gr(tX, tY, B * groups);
    switch (KC) {
      case 1:
        conv3x3_wmma<1><<<gr, 256, ldsb, stream>>>(ip, op, wpt, bias,
            Hin, Win, Hout, Wout, SOUT, CT, Cout, pad, leaky, groups);
        break;
      case 4:
        conv3x3_wmma<4><<<gr, 256, ldsb, stream>>>(ip, op, wpt, bias,
            Hin, Win, Hout, Wout, SOUT, CT, Cout, pad, leaky, groups);
        break;
      case 5:
        conv3x3_wmma<5><<<gr, 256, ldsb, stream>>>(ip, op, wpt, bias,
            Hin, Win, Hout, Wout, SOUT, CT, Cout, pad, leaky, groups);
        break;
    }
  };

  // conv0: 24(->32)ch -> 128, SAME, leaky
  conv(imap, ping, wp0, mb0, 128, 128, 128, 128, 128, 1, 8, 128, 1, 1, 1);
  const __bf16* cur = ping;
  __bf16* nxt = pong;
  for (int i = 0; i < 10; ++i) {
    conv(cur, nxt, wpm + (size_t)i * 8 * 9 * 4 * 512, mb_mid + i * 128,
         128, 128, 128, 128, 128, 4, 8, 128, 1, 1, 1);
    const __bf16* t = cur; cur = nxt; nxt = (__bf16*)t;
  }
  // mid[10] (no activation) -> channels 0..127 of the 160-ch concat buffer
  conv(cur, kbuf, wpm + (size_t)10 * 8 * 9 * 4 * 512, mb_mid + 10 * 128,
       128, 128, 128, 128, 160, 4, 8, 128, 1, 0, 1);
  // k-branch: VALID convs
  conv(kbuf, k1o, wpk0, kb0, 128, 128, 126, 126, 128, 5, 8, 128, 0, 1, 1);
  conv(k1o,  k2o, wpk1, kb1, 126, 126, 124, 124, 128, 4, 8, 128, 0, 1, 1);
  conv(k2o,  kco, wpk2, kb2, 124, 124, 122, 122, 448, 4, 28, 441, 0, 0, 4);

  apply_softmax_kernel<<<dim3((108 * 108 + 255) / 256, B), 256, 0, stream>>>(
      radm, kco, out);
}
